// MultilayerGRU_64613488001436
// MI455X (gfx1250) — matrix operand order repaired and checked by reading, not compiled, not run on hardware
//
#include <hip/hip_runtime.h>

// Problem constants (B,S,I,H,O,L) = (64,512,1024,1024,1024,2)
#define BB 64
#define SS 512
#define II 1024
#define HD 1024
#define OO 1024
#define NLAY 2
#define BMAT 2176  // staged B chunk elems: 16 groups (4 cols) * 136

typedef __attribute__((ext_vector_type(16))) __bf16 v16bf;
typedef __attribute__((ext_vector_type(8)))  float  v8f;

struct alignas(16) U4 { unsigned int x, y, z, w; };
struct alignas(8)  U2 { unsigned int x, y; };
struct alignas(16) F4 { float x, y, z, w; };
union  BPack8 { __bf16 b[8]; U4 u; };
union  BPack4 { __bf16 b[4]; U2 u; };

// ---- CDNA5 async global->LDS copy (ASYNCcnt-tracked), via inline asm -------
#if defined(__HIP_DEVICE_COMPILE__) && defined(__gfx1250__)
#define HAVE_ASYNC 1
#else
#define HAVE_ASYNC 0
#endif

__device__ __forceinline__ void async_cp16(const void* g, void* l) {
#if HAVE_ASYNC
  unsigned int lds_off = (unsigned int)(unsigned long long)(size_t)l;
  unsigned long long gaddr = (unsigned long long)(size_t)g;
  asm volatile("global_load_async_to_lds_b128 %0, %1, off"
               :
               : "v"(lds_off), "v"(gaddr)
               : "memory");
#else
  *(U4*)l = *(const U4*)g;
#endif
}

__device__ __forceinline__ void wait_async() {
#if HAVE_ASYNC
#if __has_builtin(__builtin_amdgcn_s_wait_asynccnt)
  __builtin_amdgcn_s_wait_asynccnt(0);
#else
  asm volatile("s_wait_asynccnt 0x0" ::: "memory");
#endif
#endif
}

__device__ __forceinline__ v16bf lds_ld16(const __bf16* p) {
  union { U4 u[2]; v16bf v; } t;
  t.u[0] = *(const U4*)(p);
  t.u[1] = *(const U4*)(p + 8);
  return t.v;
}

__device__ __forceinline__ v8f wmma_bf16(v16bf a, v16bf b, v8f c) {
  return __builtin_amdgcn_wmma_f32_16x16x32_bf16(false, a, false, b, (short)0, c,
                                                 false, false);
}

__device__ __forceinline__ float sigm(float x) { return 1.f / (1.f + __expf(-x)); }

// B-chunk LDS layout: conflict-free, 16B-aligned segments.
// col's 32-K region starts at (col/4)*136 + (col%4)*32 elems.
__device__ __forceinline__ int bofs(int col) {
  return ((col >> 2) * 136) + ((col & 3) << 5);
}

// Stage 64x32 fp32 A-chunk (row stride lda) into swizzled WMMA fragment order.
// lane = (row%16)+16*half; elems j: K = j%8 + 8*half + 16*(j/8).
__device__ __forceinline__ void stageA_f32(const float* __restrict__ A, int lda, int k0,
                                           __bf16* sA, int tid) {
  int row = tid >> 2;
  int c   = (tid & 3) << 3;
  const float* ap = A + (size_t)row * lda + k0 + c;
  F4 f0 = *(const F4*)(ap);
  F4 f1 = *(const F4*)(ap + 4);
  BPack8 p;
  p.b[0] = (__bf16)f0.x; p.b[1] = (__bf16)f0.y; p.b[2] = (__bf16)f0.z; p.b[3] = (__bf16)f0.w;
  p.b[4] = (__bf16)f1.x; p.b[5] = (__bf16)f1.y; p.b[6] = (__bf16)f1.z; p.b[7] = (__bf16)f1.w;
  int mt = row >> 4, lr = row & 15, half = (c >> 3) & 1, jb = (c >> 4) << 3;
  *(U4*)&sA[((mt * 32 + lr + 16 * half) << 4) + jb] = p.u;
}

// Same swizzle, bf16 source, async copy.
__device__ __forceinline__ void stageA_bf16(const __bf16* __restrict__ A, size_t lda,
                                            int k0, __bf16* sA, int tid) {
  int row = tid >> 2;
  int c   = (tid & 3) << 3;
  int mt = row >> 4, lr = row & 15, half = (c >> 3) & 1, jb = (c >> 4) << 3;
  async_cp16(A + (size_t)row * lda + k0 + c,
             &sA[((mt * 32 + lr + 16 * half) << 4) + jb]);
}

// Stage 64(cols)x32(K) chunk from TRANSPOSED weights WT[N][K] (async).
__device__ __forceinline__ void stageB_T(const __bf16* __restrict__ WT, int k0, int nBase,
                                         __bf16* sB, int tid) {
  int col  = tid >> 2;        // 0..63
  int part = (tid & 3) << 3;  // 0,8,16,24
  async_cp16(WT + (size_t)(nBase + col) * HD + k0 + part,
             &sB[bofs(col) + (((part & 8) << 1) | ((part & 16) >> 1))]);
}

// ---------------- fused z/r/gx gates (layer1 always; layer0 fallback)
// z = sig(x@Wxz + h@Whz + bz), r = sig(x@Wxr + h@Whr + br), gx = x@Wxg (raw)
// A sources are bf16 mirrors -> pure async staging, double-buffered.
__global__ __launch_bounds__(256) void gates_xh_kernel(
    const __bf16* __restrict__ xbf, int xld,
    const __bf16* __restrict__ hbf, const float* __restrict__ h,
    const __bf16* __restrict__ WxzT, const __bf16* __restrict__ WhzT,
    const __bf16* __restrict__ WxrT, const __bf16* __restrict__ WhrT,
    const __bf16* __restrict__ WxgT,
    const float* __restrict__ bz, const float* __restrict__ br,
    float* __restrict__ zout, __bf16* __restrict__ rhbf, float* __restrict__ gxout) {
  __shared__ alignas(16) __bf16 sA[2][2][2048];
  __shared__ alignas(16) __bf16 sB[2][5][BMAT];
  int tid = threadIdx.x, lane = tid & 31, wave = tid >> 5;
  int mt = wave & 3, np = wave >> 2;
  int nBase = blockIdx.x << 6;
  v8f acc[3][2];  // z, r, gx
#pragma unroll
  for (int g = 0; g < 3; ++g)
#pragma unroll
    for (int n = 0; n < 2; ++n)
#pragma unroll
      for (int i = 0; i < 8; ++i) acc[g][n][i] = 0.f;

  auto stage = [&](int k0, int buf) {
    stageB_T(WxzT, k0, nBase, sB[buf][0], tid);
    stageB_T(WhzT, k0, nBase, sB[buf][1], tid);
    stageB_T(WxrT, k0, nBase, sB[buf][2], tid);
    stageB_T(WhrT, k0, nBase, sB[buf][3], tid);
    stageB_T(WxgT, k0, nBase, sB[buf][4], tid);
    stageA_bf16(xbf, (size_t)xld, k0, sA[buf][0], tid);
    stageA_bf16(hbf, HD, k0, sA[buf][1], tid);
  };
  stage(0, 0);
  int aofs = (mt * 32 + lane) << 4;
  int lr = lane & 15, kg16 = (lane >> 4) << 4;
  int c0 = bofs(np * 32 + lr) + kg16;
  int c1 = bofs(np * 32 + 16 + lr) + kg16;
  for (int it = 0; it < 32; ++it) {
    int buf = it & 1;
    wait_async();
    __syncthreads();
    if (it + 1 < 32) stage((it + 1) << 5, buf ^ 1);
    v16bf ax = lds_ld16(&sA[buf][0][aofs]);
    v16bf ah = lds_ld16(&sA[buf][1][aofs]);
    v16bf bfr[5][2];
#pragma unroll
    for (int m = 0; m < 5; ++m) {
      bfr[m][0] = lds_ld16(&sB[buf][m][c0]);
      bfr[m][1] = lds_ld16(&sB[buf][m][c1]);
    }
#pragma unroll
    for (int ns = 0; ns < 2; ++ns) {
      acc[0][ns] = wmma_bf16(ax, bfr[0][ns], acc[0][ns]);
      acc[0][ns] = wmma_bf16(ah, bfr[1][ns], acc[0][ns]);
      acc[1][ns] = wmma_bf16(ax, bfr[2][ns], acc[1][ns]);
      acc[1][ns] = wmma_bf16(ah, bfr[3][ns], acc[1][ns]);
      acc[2][ns] = wmma_bf16(ax, bfr[4][ns], acc[2][ns]);
    }
  }
  int lh = lane >> 4;
#pragma unroll
  for (int ns = 0; ns < 2; ++ns) {
    int N = nBase + np * 32 + ns * 16 + lr;
    float bzv = bz[N], brv = br[N];
#pragma unroll
    for (int i = 0; i < 8; ++i) {
      int M = mt * 16 + lh * 8 + i;
      zout[M * HD + N]  = sigm(acc[0][ns][i] + bzv);
      rhbf[M * HD + N]  = (__bf16)(sigm(acc[1][ns][i] + brv) * h[M * HD + N]);
      gxout[M * HD + N] = acc[2][ns][i];
    }
  }
}

// ---------------- h-side only z/r (layer0 when x-projections precomputed)
__global__ __launch_bounds__(256) void gates_h_kernel(
    const __bf16* __restrict__ hbf, const float* __restrict__ h,
    const __bf16* __restrict__ WhzT, const __bf16* __restrict__ WhrT,
    const float* __restrict__ xz, const float* __restrict__ xr,
    const float* __restrict__ bz, const float* __restrict__ br,
    float* __restrict__ zout, __bf16* __restrict__ rhbf) {
  __shared__ alignas(16) __bf16 sA[2][2048];
  __shared__ alignas(16) __bf16 sB[2][2][BMAT];
  int tid = threadIdx.x, lane = tid & 31, wave = tid >> 5;
  int mt = wave & 3, np = wave >> 2;
  int nBase = blockIdx.x << 6;
  v8f acc[2][2];
#pragma unroll
  for (int g = 0; g < 2; ++g)
#pragma unroll
    for (int n = 0; n < 2; ++n)
#pragma unroll
      for (int i = 0; i < 8; ++i) acc[g][n][i] = 0.f;

  auto stage = [&](int k0, int buf) {
    stageB_T(WhzT, k0, nBase, sB[buf][0], tid);
    stageB_T(WhrT, k0, nBase, sB[buf][1], tid);
    stageA_bf16(hbf, HD, k0, sA[buf], tid);
  };
  stage(0, 0);
  int aofs = (mt * 32 + lane) << 4;
  int lr = lane & 15, kg16 = (lane >> 4) << 4;
  int c0 = bofs(np * 32 + lr) + kg16;
  int c1 = bofs(np * 32 + 16 + lr) + kg16;
  for (int it = 0; it < 32; ++it) {
    int buf = it & 1;
    wait_async();
    __syncthreads();
    if (it + 1 < 32) stage((it + 1) << 5, buf ^ 1);
    v16bf ah = lds_ld16(&sA[buf][aofs]);
    v16bf bfr[2][2];
#pragma unroll
    for (int m = 0; m < 2; ++m) {
      bfr[m][0] = lds_ld16(&sB[buf][m][c0]);
      bfr[m][1] = lds_ld16(&sB[buf][m][c1]);
    }
#pragma unroll
    for (int ns = 0; ns < 2; ++ns) {
      acc[0][ns] = wmma_bf16(ah, bfr[0][ns], acc[0][ns]);
      acc[1][ns] = wmma_bf16(ah, bfr[1][ns], acc[1][ns]);
    }
  }
  int lh = lane >> 4;
#pragma unroll
  for (int ns = 0; ns < 2; ++ns) {
    int N = nBase + np * 32 + ns * 16 + lr;
    float bzv = bz[N], brv = br[N];
#pragma unroll
    for (int i = 0; i < 8; ++i) {
      int M = mt * 16 + lh * 8 + i;
      zout[M * HD + N] = sigm(acc[0][ns][i] + xz[M * HD + N] + bzv);
      rhbf[M * HD + N] =
          (__bf16)(sigm(acc[1][ns][i] + xr[M * HD + N] + brv) * h[M * HD + N]);
    }
  }
}

// ---------------- g gate + state update (single matmul: rh@Whg)
__global__ __launch_bounds__(256) void gcell_kernel(
    const __bf16* __restrict__ rhbf, const __bf16* __restrict__ WhgT,
    const float* __restrict__ gx, const float* __restrict__ bg,
    const float* __restrict__ z, float* __restrict__ h,
    __bf16* __restrict__ hbfout, __bf16* __restrict__ harch) {
  __shared__ alignas(16) __bf16 sA[2][2048];
  __shared__ alignas(16) __bf16 sB[2][BMAT];
  int tid = threadIdx.x, lane = tid & 31, wave = tid >> 5;
  int mt = wave & 3, np = wave >> 2;
  int nBase = blockIdx.x << 6;
  v8f acc[2];
#pragma unroll
  for (int n = 0; n < 2; ++n)
#pragma unroll
    for (int i = 0; i < 8; ++i) acc[n][i] = 0.f;

  auto stage = [&](int k0, int buf) {
    stageB_T(WhgT, k0, nBase, sB[buf], tid);
    stageA_bf16(rhbf, HD, k0, sA[buf], tid);
  };
  stage(0, 0);
  int aofs = (mt * 32 + lane) << 4;
  int lr = lane & 15, kg16 = (lane >> 4) << 4;
  int c0 = bofs(np * 32 + lr) + kg16;
  int c1 = bofs(np * 32 + 16 + lr) + kg16;
  for (int it = 0; it < 32; ++it) {
    int buf = it & 1;
    wait_async();
    __syncthreads();
    if (it + 1 < 32) stage((it + 1) << 5, buf ^ 1);
    v16bf ar = lds_ld16(&sA[buf][aofs]);
    v16bf b0 = lds_ld16(&sB[buf][c0]);
    v16bf b1 = lds_ld16(&sB[buf][c1]);
    acc[0] = wmma_bf16(ar, b0, acc[0]);
    acc[1] = wmma_bf16(ar, b1, acc[1]);
  }
  int lh = lane >> 4;
#pragma unroll
  for (int ns = 0; ns < 2; ++ns) {
    int N = nBase + np * 32 + ns * 16 + lr;
    float bgv = bg[N];
#pragma unroll
    for (int i = 0; i < 8; ++i) {
      int M = mt * 16 + lh * 8 + i;
      float gv = tanhf(acc[ns][i] + gx[M * HD + N] + bgv);
      float zv = z[M * HD + N];
      float hn = zv * h[M * HD + N] + (1.f - zv) * gv;
      h[M * HD + N] = hn;
      hbfout[M * HD + N] = (__bf16)hn;
      if (harch) harch[M * HD + N] = (__bf16)hn;
    }
  }
}

// ---------------- bulk layer0 x-projections (throughput GEMM, occupancy-bound)
__global__ __launch_bounds__(256) void xproj_kernel(
    const float* __restrict__ input,
    const __bf16* __restrict__ WxzT, const __bf16* __restrict__ WxrT,
    const __bf16* __restrict__ WxgT,
    float* __restrict__ xz, float* __restrict__ xr, float* __restrict__ xg) {
  __shared__ alignas(16) __bf16 sA[2048];
  __shared__ alignas(16) __bf16 sB[3][BMAT];
  int tid = threadIdx.x, lane = tid & 31, wave = tid >> 5;
  int mt = wave & 3, np = wave >> 2;
  int nBase = blockIdx.x << 6;
  int rowBase = blockIdx.y << 6;
  int row = tid >> 2, c = (tid & 3) << 3;
  int R = rowBase + row, b = R & (BB - 1), t = R >> 6;
  const float* arow = input + ((size_t)b * SS + t) * II;  // gathered A row
  v8f acc[3][2];
#pragma unroll
  for (int g = 0; g < 3; ++g)
#pragma unroll
    for (int n = 0; n < 2; ++n)
#pragma unroll
      for (int i = 0; i < 8; ++i) acc[g][n][i] = 0.f;

  int aofs = (mt * 32 + lane) << 4;
  int lr = lane & 15, kg16 = (lane >> 4) << 4;
  int c0 = bofs(np * 32 + lr) + kg16;
  int c1 = bofs(np * 32 + 16 + lr) + kg16;
  for (int k0 = 0; k0 < HD; k0 += 32) {
    stageB_T(WxzT, k0, nBase, sB[0], tid);
    stageB_T(WxrT, k0, nBase, sB[1], tid);
    stageB_T(WxgT, k0, nBase, sB[2], tid);
    {
      const float* ap = arow + k0 + c;
      if (k0 + 32 < HD) __builtin_prefetch(ap + 32, 0, 1);
      F4 f0 = *(const F4*)(ap);
      F4 f1 = *(const F4*)(ap + 4);
      BPack8 p;
      p.b[0] = (__bf16)f0.x; p.b[1] = (__bf16)f0.y; p.b[2] = (__bf16)f0.z; p.b[3] = (__bf16)f0.w;
      p.b[4] = (__bf16)f1.x; p.b[5] = (__bf16)f1.y; p.b[6] = (__bf16)f1.z; p.b[7] = (__bf16)f1.w;
      int mt2 = row >> 4, lr2 = row & 15, half = (c >> 3) & 1, jb = (c >> 4) << 3;
      *(U4*)&sA[((mt2 * 32 + lr2 + 16 * half) << 4) + jb] = p.u;
    }
    wait_async();
    __syncthreads();
    v16bf a = lds_ld16(&sA[aofs]);
    v16bf bfr[3][2];
#pragma unroll
    for (int m = 0; m < 3; ++m) {
      bfr[m][0] = lds_ld16(&sB[m][c0]);
      bfr[m][1] = lds_ld16(&sB[m][c1]);
    }
#pragma unroll
    for (int ns = 0; ns < 2; ++ns) {
      acc[0][ns] = wmma_bf16(a, bfr[0][ns], acc[0][ns]);
      acc[1][ns] = wmma_bf16(a, bfr[1][ns], acc[1][ns]);
      acc[2][ns] = wmma_bf16(a, bfr[2][ns], acc[2][ns]);
    }
    __syncthreads();
  }
  int lh = lane >> 4;
#pragma unroll
  for (int ns = 0; ns < 2; ++ns) {
    int N = nBase + np * 32 + ns * 16 + lr;
#pragma unroll
    for (int i = 0; i < 8; ++i) {
      size_t o = (size_t)(rowBase + mt * 16 + lh * 8 + i) * HD + N;
      xz[o] = acc[0][ns][i];
      xr[o] = acc[1][ns][i];
      xg[o] = acc[2][ns][i];
    }
  }
}

// ---------------- deferred output projection Y = H1all @ Why + by
__global__ __launch_bounds__(256) void y_gemm_kernel(
    const __bf16* __restrict__ A, const __bf16* __restrict__ WT,
    const float* __restrict__ by, float* __restrict__ out) {
  __shared__ alignas(16) __bf16 sA[2048];
  __shared__ alignas(16) __bf16 sB[BMAT];
  int tid = threadIdx.x, lane = tid & 31, wave = tid >> 5;
  int mt = wave & 3, np = wave >> 2;
  int nBase = blockIdx.x << 6;
  int rowBase = blockIdx.y << 6;
  v8f acc[2];
#pragma unroll
  for (int n = 0; n < 2; ++n)
#pragma unroll
    for (int i = 0; i < 8; ++i) acc[n][i] = 0.f;

  int aofs = (mt * 32 + lane) << 4;
  int lr = lane & 15, kg16 = (lane >> 4) << 4;
  int c0 = bofs(np * 32 + lr) + kg16;
  int c1 = bofs(np * 32 + 16 + lr) + kg16;
  for (int k0 = 0; k0 < HD; k0 += 32) {
    stageB_T(WT, k0, nBase, sB, tid);
    stageA_bf16(A + (size_t)rowBase * HD, HD, k0, sA, tid);
    wait_async();
    __syncthreads();
    v16bf a = lds_ld16(&sA[aofs]);
    v16bf b0 = lds_ld16(&sB[c0]);
    v16bf b1 = lds_ld16(&sB[c1]);
    acc[0] = wmma_bf16(a, b0, acc[0]);
    acc[1] = wmma_bf16(a, b1, acc[1]);
    __syncthreads();
  }
  int lh = lane >> 4;
#pragma unroll
  for (int ns = 0; ns < 2; ++ns) {
    int N = nBase + np * 32 + ns * 16 + lr;
    float bv = by[N];
#pragma unroll
    for (int i = 0; i < 8; ++i) {
      int M = rowBase + mt * 16 + lh * 8 + i;  // = t*BB + b
      int b = M & (BB - 1);
      int t = M >> 6;
      out[((size_t)b * SS + t) * OO + N] = acc[ns][i] + bv;
    }
  }
}

// ---------------- utilities
__global__ __launch_bounds__(256) void cvtT_kernel(const float* __restrict__ src,
                                                   __bf16* __restrict__ dst) {
  __shared__ float tile[32][33];
  int bx = blockIdx.x << 5, by = blockIdx.y << 5;  // bx: n, by: k
  int tx = threadIdx.x & 31, ty = threadIdx.x >> 5;
#pragma unroll
  for (int i = ty; i < 32; i += 8)
    tile[i][tx] = src[(size_t)(by + i) * HD + bx + tx];
  __syncthreads();
#pragma unroll
  for (int i = ty; i < 32; i += 8)
    dst[(size_t)(bx + i) * HD + by + tx] = (__bf16)tile[tx][i];
}

__global__ void cvt4_kernel(const float* __restrict__ src, __bf16* __restrict__ dst,
                            int n4) {
  int i = blockIdx.x * blockDim.x + threadIdx.x;
  if (i < n4) {
    F4 f = *(const F4*)(src + (size_t)i * 4);
    BPack4 p;
    p.b[0] = (__bf16)f.x; p.b[1] = (__bf16)f.y; p.b[2] = (__bf16)f.z; p.b[3] = (__bf16)f.w;
    *(U2*)(dst + (size_t)i * 4) = p.u;
  }
}

__global__ void zero_f32_kernel(float* __restrict__ p, int n) {
  int i = blockIdx.x * blockDim.x + threadIdx.x;
  if (i < n) p[i] = 0.f;
}

__global__ void zero_bf16_kernel(__bf16* __restrict__ p, int n) {
  int i = blockIdx.x * blockDim.x + threadIdx.x;
  if (i < n) p[i] = (__bf16)0.f;
}

__global__ void hid_copy_kernel(const float* __restrict__ h0,
                                const float* __restrict__ h1,
                                float* __restrict__ out2) {
  int i = blockIdx.x * blockDim.x + threadIdx.x;
  if (i < BB * HD) {
    int b = i >> 10, k = i & 1023;
    out2[b * (NLAY * HD) + k]      = h0[i];
    out2[b * (NLAY * HD) + HD + k] = h1[i];
  }
}

extern "C" void kernel_launch(void* const* d_in, const int* in_sizes, int n_in,
                              void* d_out, int out_size, void* d_ws, size_t ws_size,
                              hipStream_t stream) {
  const float* input = (const float*)d_in[0];
  const float* Wxz = (const float*)d_in[1];
  const float* Whz = (const float*)d_in[2];
  const float* bz  = (const float*)d_in[3];
  const float* Wxr = (const float*)d_in[4];
  const float* Whr = (const float*)d_in[5];
  const float* br  = (const float*)d_in[6];
  const float* Wxg = (const float*)d_in[7];
  const float* Whg = (const float*)d_in[8];
  const float* bg  = (const float*)d_in[9];
  const float* Why = (const float*)d_in[10];
  const float* by  = (const float*)d_in[11];

  char* ws = (char*)d_ws;
  size_t off = 0;
  auto salloc = [&](size_t bytes) -> void* {
    void* p = ws + off;
    off += (bytes + 255) & ~(size_t)255;
    return p;
  };
  const size_t LW  = (size_t)II * HD;      // per-layer weight elems
  const size_t STP = (size_t)BB * HD;      // per-step activation elems
  __bf16* wxzT = (__bf16*)salloc(NLAY * LW * 2);
  __bf16* whzT = (__bf16*)salloc(NLAY * LW * 2);
  __bf16* wxrT = (__bf16*)salloc(NLAY * LW * 2);
  __bf16* whrT = (__bf16*)salloc(NLAY * LW * 2);
  __bf16* wxgT = (__bf16*)salloc(NLAY * LW * 2);
  __bf16* whgT = (__bf16*)salloc(NLAY * LW * 2);
  __bf16* whyT = (__bf16*)salloc(LW * 2);
  float*  h0    = (float*)salloc(STP * 4);
  float*  h1    = (float*)salloc(STP * 4);
  float*  zbuf  = (float*)salloc(STP * 4);
  float*  gxbuf = (float*)salloc(STP * 4);
  __bf16* hbf0  = (__bf16*)salloc(STP * 2);
  __bf16* hbf1  = (__bf16*)salloc(STP * 2);
  __bf16* rhbf  = (__bf16*)salloc(STP * 2);
  __bf16* xbf   = (__bf16*)salloc((size_t)BB * SS * II * 2);
  __bf16* h1all = (__bf16*)salloc((size_t)SS * STP * 2);
  float* xzp = (float*)salloc((size_t)SS * STP * 4);
  float* xrp = (float*)salloc((size_t)SS * STP * 4);
  float* xgp = (float*)salloc((size_t)SS * STP * 4);
  const bool use_pre = (off <= ws_size);

  // weight transpose+convert (13 1024x1024 blocks)
  dim3 tg(32, 32);
  for (int l = 0; l < NLAY; ++l) {
    cvtT_kernel<<<tg, 256, 0, stream>>>(Wxz + l * LW, wxzT + l * LW);
    cvtT_kernel<<<tg, 256, 0, stream>>>(Whz + l * LW, whzT + l * LW);
    cvtT_kernel<<<tg, 256, 0, stream>>>(Wxr + l * LW, wxrT + l * LW);
    cvtT_kernel<<<tg, 256, 0, stream>>>(Whr + l * LW, whrT + l * LW);
    cvtT_kernel<<<tg, 256, 0, stream>>>(Wxg + l * LW, wxgT + l * LW);
    cvtT_kernel<<<tg, 256, 0, stream>>>(Whg + l * LW, whgT + l * LW);
  }
  cvtT_kernel<<<tg, 256, 0, stream>>>(Why, whyT);
  zero_f32_kernel<<<(BB * HD + 255) / 256, 256, 0, stream>>>(h0, BB * HD);
  zero_f32_kernel<<<(BB * HD + 255) / 256, 256, 0, stream>>>(h1, BB * HD);
  zero_bf16_kernel<<<(BB * HD + 255) / 256, 256, 0, stream>>>(hbf0, BB * HD);
  zero_bf16_kernel<<<(BB * HD + 255) / 256, 256, 0, stream>>>(hbf1, BB * HD);

  if (use_pre) {  // bulk x-projections for layer0
    dim3 xg2(OO / 64, (SS * BB) / 64);
    xproj_kernel<<<xg2, 256, 0, stream>>>(input, wxzT, wxrT, wxgT, xzp, xrp, xgp);
  } else {        // bf16 mirror of the input for async A staging
    int n4 = (BB * SS * II) / 4;
    cvt4_kernel<<<(n4 + 255) / 256, 256, 0, stream>>>(input, xbf, n4);
  }

  for (int t = 0; t < SS; ++t) {
    if (use_pre) {
      gates_h_kernel<<<HD / 64, 256, 0, stream>>>(
          hbf0, h0, whzT, whrT, xzp + t * STP, xrp + t * STP, bz, br, zbuf, rhbf);
      gcell_kernel<<<HD / 64, 256, 0, stream>>>(
          rhbf, whgT, xgp + t * STP, bg, zbuf, h0, hbf0, (__bf16*)nullptr);
    } else {
      gates_xh_kernel<<<HD / 64, 256, 0, stream>>>(
          xbf + (size_t)t * II, SS * II, hbf0, h0, wxzT, whzT, wxrT, whrT, wxgT,
          bz, br, zbuf, rhbf, gxbuf);
      gcell_kernel<<<HD / 64, 256, 0, stream>>>(
          rhbf, whgT, gxbuf, bg, zbuf, h0, hbf0, (__bf16*)nullptr);
    }
    // layer 1 (input = h0 / hbf0)
    gates_xh_kernel<<<HD / 64, 256, 0, stream>>>(
        hbf0, HD, hbf1, h1, wxzT + LW, whzT + LW, wxrT + LW, whrT + LW, wxgT + LW,
        bz + HD, br + HD, zbuf, rhbf, gxbuf);
    gcell_kernel<<<HD / 64, 256, 0, stream>>>(
        rhbf, whgT + LW, gxbuf, bg + HD, zbuf, h1, hbf1, h1all + t * STP);
  }

  dim3 yg(OO / 64, (SS * BB) / 64);
  y_gemm_kernel<<<yg, 256, 0, stream>>>(h1all, whyT, by, (float*)d_out);
  hid_copy_kernel<<<(BB * HD + 255) / 256, 256, 0, stream>>>(
      h0, h1, (float*)d_out + (size_t)BB * SS * OO);
}
